// ODEBlock_70815420776451
// MI455X (gfx1250) — matrix operand order repaired and checked by reading, not compiled, not run on hardware
//
#include <hip/hip_runtime.h>
#include <stdint.h>

typedef __attribute__((ext_vector_type(16))) _Float16 v16h;
typedef __attribute__((ext_vector_type(2)))  __fp16   v2hp;   // cvt_pkrtz result type
typedef __attribute__((ext_vector_type(8)))  float    v8f;

#define WAVES 8
#define ROWS_PER_BLOCK (WAVES * 16)
#define RK_H 0.05f

// ---------- helpers ----------

// swap lane halves (lanes 0-15 <-> 16-31) via ds_swizzle SWAPX16 (xor=0x10,and=0x1f)
__device__ __forceinline__ uint32_t swap16(uint32_t x) {
  return (uint32_t)__builtin_amdgcn_ds_swizzle((int)x, 0x401F);
}

// split two f32 into packed f16 hi-parts and packed f16 lo-residuals.
// hi uses RTZ (one v_cvt_pk_rtz_f16_f32); the residual is computed from the
// actual hi value, so hi's rounding mode does not affect combined accuracy.
__device__ __forceinline__ void split2(float x0, float x1, uint32_t& hi, uint32_t& lo) {
  v2hp h = __builtin_amdgcn_cvt_pkrtz(x0, x1);
  float r0 = x0 - (float)h.x;
  float r1 = x1 - (float)h.y;
  v2hp l = __builtin_amdgcn_cvt_pkrtz(r0, r1);
  union { v2hp h; uint32_t u; } a, b;
  a.h = h; b.h = l;
  hi = a.u; lo = b.u;
}

__device__ __forceinline__ float tanh_fast(float x) {
#if __has_builtin(__builtin_amdgcn_tanhf)
  return __builtin_amdgcn_tanhf(x);
#elif __has_builtin(__builtin_amdgcn_tanh_f32)
  return __builtin_amdgcn_tanh_f32(x);
#else
  float ax = __builtin_fabsf(x);
  float t  = __builtin_amdgcn_exp2f(ax * -2.885390081777927f);
  float r  = (1.0f - t) * __builtin_amdgcn_rcpf(1.0f + t);
  return __builtin_copysignf(r, x);
#endif
}

__device__ __forceinline__ v16h as_v16h(const uint32_t u[8]) {
  union { uint32_t u[8]; v16h h; } U;
#pragma unroll
  for (int i = 0; i < 8; ++i) U.u[i] = u[i];
  return U.h;
}

// load a 16x32 f16 A-fragment (8 dwords per lane) from LDS
__device__ __forceinline__ v16h loadA(const uint32_t* p) {
  union { uint4 q[2]; v16h h; } U;
  U.q[0] = ((const uint4*)p)[0];
  U.q[1] = ((const uint4*)p)[1];
  return U.h;
}

// Convert two C/D f32 tiles (rows K = 0..31 of this chunk, cols N = batch) into a
// 32x16 f16 B-fragment (hi and lo planes).  Dest lane l, dword j (slots 2j,2j+1):
//   j<4 : low half <- T0 pair(2j) own lane ; high half <- T1 pair(2j) from lane-16
//   j>=4: low half <- T0 pair(2j) from lane+16 ; high half <- T1 pair(2j) own lane
// One swizzle per pair: swap the value the *other* half needs.
__device__ __forceinline__ void buildB(const v8f& T0, const v8f& T1, bool lowhalf,
                                       uint32_t bh[8], uint32_t bl[8]) {
#pragma unroll
  for (int j = 0; j < 4; ++j) {
    uint32_t h0, l0, h1, l1;
    split2(T0[2 * j], T0[2 * j + 1], h0, l0);
    split2(T1[2 * j], T1[2 * j + 1], h1, l1);
    uint32_t selh = lowhalf ? h1 : h0;   // value the other half wants
    uint32_t sell = lowhalf ? l1 : l0;
    uint32_t ssh = swap16(selh);
    uint32_t ssl = swap16(sell);
    bh[j]     = lowhalf ? h0  : ssh;
    bh[j + 4] = lowhalf ? ssh : h1;
    bl[j]     = lowhalf ? l0  : ssl;
    bl[j + 4] = lowhalf ? ssl : l1;
  }
}

// one MLP eval: kk = W2^T . tanh(W1^T . z + b1) + b2    (all transposed, N=16 batch)
// Interleaved: hidden tile pair -> tanh -> B-fragment -> layer-2 WMMAs, so only
// 2 hidden tiles (16 VGPRs) are ever live.  sched_barrier(0) per iteration keeps
// the scheduler's A-fragment lookahead window to one iteration (no spills).
__device__ __forceinline__ void feval(const uint32_t (&bf)[2][2][8], v8f (&kk)[4],
                                      const uint32_t (&sA)[2][64][32][8],
                                      const float* sB1, const float* sB2,
                                      int lane, int hl8, bool lowhalf) {
#pragma unroll
  for (int t = 0; t < 4; ++t) {
    float4 ba = *(const float4*)&sB2[16 * t + hl8];
    float4 bb = *(const float4*)&sB2[16 * t + hl8 + 4];
    kk[t] = v8f{ba.x, ba.y, ba.z, ba.w, bb.x, bb.y, bb.z, bb.w};
  }
  __builtin_amdgcn_sched_barrier(0);

#pragma unroll
  for (int c = 0; c < 8; ++c) {
    v8f T[2];
#pragma unroll
    for (int i = 0; i < 2; ++i) {
      const int mt = 2 * c + i;
      float4 ba = *(const float4*)&sB1[16 * mt + hl8];
      float4 bb = *(const float4*)&sB1[16 * mt + hl8 + 4];
      v8f acc = {ba.x, ba.y, ba.z, ba.w, bb.x, bb.y, bb.z, bb.w};
#pragma unroll
      for (int kc = 0; kc < 2; ++kc) {
        v16h ahi = loadA(&sA[0][mt * 2 + kc][lane][0]);
        v16h alo = loadA(&sA[1][mt * 2 + kc][lane][0]);
        v16h bhi = as_v16h(bf[kc][0]);
        v16h blo = as_v16h(bf[kc][1]);
        acc = __builtin_amdgcn_wmma_f32_16x16x32_f16(false, ahi, false, bhi, (short)0, acc, false, false);
        acc = __builtin_amdgcn_wmma_f32_16x16x32_f16(false, ahi, false, blo, (short)0, acc, false, false);
        acc = __builtin_amdgcn_wmma_f32_16x16x32_f16(false, alo, false, bhi, (short)0, acc, false, false);
      }
#pragma unroll
      for (int v = 0; v < 8; ++v) acc[v] = tanh_fast(acc[v]);
      T[i] = acc;
    }

    uint32_t bhw[8], blw[8];
    buildB(T[0], T[1], lowhalf, bhw, blw);
    v16h Bhi = as_v16h(bhw), Blo = as_v16h(blw);
#pragma unroll
    for (int t = 0; t < 4; ++t) {
      v16h ahi = loadA(&sA[0][32 + t * 8 + c][lane][0]);
      v16h alo = loadA(&sA[1][32 + t * 8 + c][lane][0]);
      kk[t] = __builtin_amdgcn_wmma_f32_16x16x32_f16(false, ahi, false, Bhi, (short)0, kk[t], false, false);
      kk[t] = __builtin_amdgcn_wmma_f32_16x16x32_f16(false, ahi, false, Blo, (short)0, kk[t], false, false);
      kk[t] = __builtin_amdgcn_wmma_f32_16x16x32_f16(false, alo, false, Bhi, (short)0, kk[t], false, false);
    }
    __builtin_amdgcn_sched_barrier(0);
  }
}

// ---------- kernel ----------

// LDS (128KB/WG) limits us to 2 WGs per WGP = 4 waves per SIMD; declaring that
// caps the VGPR budget at 256 so we avoid VGPR-MSB mode switching entirely.
__global__ __launch_bounds__(256, 4) void node_rk4_wmma(
    const float* __restrict__ x, const float* __restrict__ W1,
    const float* __restrict__ b1, const float* __restrict__ W2,
    const float* __restrict__ b2, float* __restrict__ out) {
  // A-fragments of W1^T (frags 0..31: [mt 0..15][kc 0..1]) and
  // W2^T (frags 32..63: [mt2 0..3][kc2 0..7]); plane 0 = f16 hi, plane 1 = f16 lo.
  __shared__ __align__(16) uint32_t sA[2][64][32][8];  // 128 KB
  __shared__ __align__(16) float sB1[256];
  __shared__ __align__(16) float sB2[64];

  const int tid = threadIdx.x;
  const int wave = tid >> 5;
  const int lane = tid & 31;
  const int lm = lane & 15;
  const int hl8 = (lane >> 4) * 8;
  const bool lowhalf = (lane < 16);

  // ---- one-time weight prep: W^T into A-fragment layout, hi/lo split ----
  // A 16x32 f16 layout as used here (contiguous k, matching buildB's B): lane = M + 16*(K/16); slot s -> K = s + 16*(lane/16)
#pragma unroll 1
  for (int u = wave; u < 64; u += WAVES) {
    const bool l1 = (u < 32);
    const int mt = l1 ? (u >> 1) : ((u - 32) >> 3);
    const int kc = l1 ? (u & 1) : ((u - 32) & 7);
    const float* W = l1 ? W1 : W2;
    const int ldw = l1 ? 256 : 64;   // row length of original (D,HID) / (HID,D) matrix
    const int m = 16 * mt + lm;      // output-dim index of W^T
#pragma unroll
    for (int p = 0; p < 8; ++p) {
      const int kk0 = 32 * kc + 2 * p + 2 * hl8;
      float w0 = W[(size_t)kk0 * ldw + m];
      float w1 = W[(size_t)(kk0 + 1) * ldw + m];
      uint32_t hi, lo;
      split2(w0, w1, hi, lo);
      sA[0][u][lane][p] = hi;
      sA[1][u][lane][p] = lo;
    }
  }
  sB1[tid] = b1[tid];
  if (tid < 64) sB2[tid] = b2[tid];
  __syncthreads();

  // ---- load y^T (C/D layout: lane holds feat = v + hl8 + 16t, batch = lane%16) ----
  const size_t row = (size_t)blockIdx.x * ROWS_PER_BLOCK + (size_t)wave * 16 + lm;
  const float* xr = x + row * 64;
  v8f yv[4];
#pragma unroll
  for (int t = 0; t < 4; ++t) {
    float4 a = *(const float4*)(xr + hl8 + 16 * t);
    float4 b = *(const float4*)(xr + hl8 + 16 * t + 4);
    yv[t] = v8f{a.x, a.y, a.z, a.w, b.x, b.y, b.z, b.w};
  }

  v8f kk[4];
#pragma unroll
  for (int t = 0; t < 4; ++t) kk[t] = v8f{0, 0, 0, 0, 0, 0, 0, 0};

  // ---- 20 RK4 steps, everything in registers ----
#pragma unroll 1
  for (int step = 0; step < 20; ++step) {
    v8f rk[4];
#pragma unroll
    for (int t = 0; t < 4; ++t) rk[t] = v8f{0, 0, 0, 0, 0, 0, 0, 0};

#pragma unroll 1
    for (int sub = 0; sub < 4; ++sub) {
      const float cin = (sub == 0) ? 0.0f : ((sub == 3) ? RK_H : 0.5f * RK_H);
      const float wgt = (sub == 1 || sub == 2) ? 2.0f : 1.0f;

      uint32_t bf[2][2][8];
#pragma unroll
      for (int c = 0; c < 2; ++c) {
        v8f t0, t1;
#pragma unroll
        for (int v = 0; v < 8; ++v) {
          t0[v] = yv[2 * c][v] + cin * kk[2 * c][v];
          t1[v] = yv[2 * c + 1][v] + cin * kk[2 * c + 1][v];
        }
        buildB(t0, t1, lowhalf, bf[c][0], bf[c][1]);
      }
      __builtin_amdgcn_sched_barrier(0);

      feval(bf, kk, sA, sB1, sB2, lane, hl8, lowhalf);

#pragma unroll
      for (int t = 0; t < 4; ++t)
#pragma unroll
        for (int v = 0; v < 8; ++v) rk[t][v] += wgt * kk[t][v];
    }

#pragma unroll
    for (int t = 0; t < 4; ++t)
#pragma unroll
      for (int v = 0; v < 8; ++v) yv[t][v] += (RK_H / 6.0f) * rk[t][v];
  }

  // ---- store y ----
  float* orow = out + row * 64;
#pragma unroll
  for (int t = 0; t < 4; ++t) {
    float4 a = {yv[t][0], yv[t][1], yv[t][2], yv[t][3]};
    float4 b = {yv[t][4], yv[t][5], yv[t][6], yv[t][7]};
    *(float4*)(orow + hl8 + 16 * t) = a;
    *(float4*)(orow + hl8 + 16 * t + 4) = b;
  }
}

// ---------- launch ----------

extern "C" void kernel_launch(void* const* d_in, const int* in_sizes, int n_in,
                              void* d_out, int out_size, void* d_ws, size_t ws_size,
                              hipStream_t stream) {
  (void)in_sizes; (void)n_in; (void)out_size; (void)d_ws; (void)ws_size;
  const float* x  = (const float*)d_in[0];
  const float* W1 = (const float*)d_in[1];
  const float* b1 = (const float*)d_in[2];
  const float* W2 = (const float*)d_in[3];
  const float* b2 = (const float*)d_in[4];
  float* out = (float*)d_out;

  dim3 grid(131072 / ROWS_PER_BLOCK);  // 1024 blocks
  dim3 block(256);                     // 8 waves, 16 rows each
  node_rk4_wmma<<<grid, block, 0, stream>>>(x, W1, b1, W2, b2, out);
}
